// PotentialNeuralNet_7971459302022
// MI455X (gfx1250) — matrix operand order repaired and checked by reading, not compiled, not run on hardware
//
#include <hip/hip_runtime.h>
#include <hip/hip_bf16.h>

// ---------------- problem constants (match reference) ----------------
#define NO_   128      // descriptor dim
#define H_    166      // hidden dim
#define HP_   192      // H padded to multiple of 32 (WMMA K)
#define HT_   176      // H padded to multiple of 16 (forward N-tiles = 11)
#define S_    4        // species / experts
#define NTA_  500000   // atoms
#define NC_   4096     // crystals
#define MT_   128      // atoms per block (8 waves x 16-row M tiles)

typedef __attribute__((ext_vector_type(16))) _Float16 v16h;
typedef __attribute__((ext_vector_type(8)))  float    v8f;

// ---------------- LDS budget (dynamic shared) ----------------
static constexpr int SM_HALVES = NO_*HP_ + HP_*NO_ + MT_*NO_ + MT_*NO_ + MT_*HP_;
static constexpr int SM_FLOATS = MT_*3 + MT_*9 + 3*NO_ + 9*NO_ + HP_ + HP_;
static constexpr int SM_INTS   = 2*MT_;
static constexpr int SMEM_BYTES = SM_HALVES*2 + SM_FLOATS*4 + SM_INTS*4; // ~223 KB < 320 KB/WGP

// fast sigmoid: v_exp_f32 + v_rcp_f32 (no IEEE divide sequence)
__device__ __forceinline__ float fast_sigmoid(float z) {
  return __builtin_amdgcn_rcpf(1.0f + __expf(-z));
}

// ---------------- tiny setup kernels ----------------
__global__ void k_init(int* counts, unsigned long long* esum) {
  int i = blockIdx.x * 256 + threadIdx.x;
  if (i < NC_) esum[i] = 0ull;
  if (blockIdx.x == 0 && threadIdx.x < S_) counts[threadIdx.x] = 0;
}

__global__ void k_count(const int* __restrict__ sym, int* counts) {
  __shared__ int h[S_];
  int tid = threadIdx.x;
  if (tid < S_) h[tid] = 0;
  __syncthreads();
  int i = blockIdx.x * 256 + tid;
  if (i < NTA_) atomicAdd(&h[sym[i]], 1);
  __syncthreads();
  if (tid < S_) atomicAdd(&counts[tid], h[tid]);
}

__global__ void k_scan(const int* counts, int* offsets, int* cursors, int* tileStart) {
  if (threadIdx.x == 0) {
    int off = 0, ts = 0;
    offsets[0] = 0; tileStart[0] = 0;
    for (int s = 0; s < S_; ++s) {
      cursors[s] = off;
      off += counts[s];
      ts  += (counts[s] + MT_ - 1) / MT_;
      offsets[s + 1]   = off;
      tileStart[s + 1] = ts;
    }
  }
}

__global__ void k_scatter(const int* __restrict__ sym, int* cursors, int* sortedIdx) {
  int i = blockIdx.x * 256 + threadIdx.x;
  if (i < NTA_) {
    int slot = atomicAdd(&cursors[sym[i]], 1);
    sortedIdx[slot] = i;
  }
}

__global__ void k_final(const unsigned long long* esum, float* energy) {
  int i = blockIdx.x * 256 + threadIdx.x;
  if (i < NC_)
    energy[i] = (float)((double)(long long)esum[i] * (1.0 / 4294967296.0));
}

// ---------------- fused forward+backward expert kernel ----------------
__global__ __launch_bounds__(256, 1)
void k_main(const float* __restrict__ positions, const float* __restrict__ cells,
            const int* __restrict__ crystalidx,
            const float* __restrict__ Wd, const float* __restrict__ Wc,
            const float* __restrict__ W1, const float* __restrict__ b1,
            const float* __restrict__ W2, const float* __restrict__ b2,
            const int* __restrict__ offsets, const int* __restrict__ tileStart,
            const int* __restrict__ sortedIdx,
            float* __restrict__ energies, unsigned long long* __restrict__ esum,
            float* __restrict__ forces) {
  int t = blockIdx.x;
  if (t >= tileStart[S_]) return;

  int s = 0;
#pragma unroll
  for (int k = 1; k < S_; ++k) if (t >= tileStart[k]) s = k;

  const int base   = offsets[s] + (t - tileStart[s]) * MT_;
  const int segEnd = offsets[s + 1];
  const int tid    = threadIdx.x;

  extern __shared__ char smem[];
  _Float16* W1hB  = (_Float16*)smem;            // [NO][HP]  backward B (W1^T reads)
  _Float16* W1tB  = W1hB + NO_ * HP_;           // [HP][NO]  forward  B (contiguous K)
  _Float16* featB = W1tB + HP_ * NO_;           // [MT][NO]
  _Float16* cosB  = featB + MT_ * NO_;          // [MT][NO]
  _Float16* gzB   = cosB  + MT_ * NO_;          // [MT][HP]
  float*    posB  = (float*)(gzB + MT_ * HP_);  // [MT][3]
  float*    cellB = posB  + MT_ * 3;            // [MT][9]
  float*    WdB   = cellB + MT_ * 9;            // [3][NO]
  float*    WcB   = WdB   + 3 * NO_;            // [9][NO]
  float*    b1B   = WcB   + 9 * NO_;            // [HP]
  float*    W2B   = b1B   + HP_;                // [HP]
  int*      aidxB = (int*)(W2B + HP_);          // [MT]
  int*      cidxB = aidxB + MT_;                // [MT]

  // ---- P1: gather atom indices / crystal ids / positions ----
  for (int i = tid; i < MT_; i += 256) {
    int p  = base + i;
    int ai = (p < segEnd) ? sortedIdx[p] : -1;
    aidxB[i] = ai;
    cidxB[i] = (ai >= 0) ? crystalidx[ai] : 0;
#pragma unroll
    for (int k = 0; k < 3; ++k)
      posB[i * 3 + k] = (ai >= 0) ? positions[ai * 3 + k] : 0.0f;
  }
  __syncthreads();

  // ---- P2: stage weights + per-atom cell rows; zero gz pad columns ----
  for (int i = tid; i < MT_ * 9; i += 256) {
    int a = i / 9, q = i % 9;
    cellB[i] = (aidxB[a] >= 0) ? cells[cidxB[a] * 9 + q] : 0.0f;
  }
  for (int i = tid; i < 3 * NO_; i += 256) WdB[i] = Wd[i];
  for (int i = tid; i < 9 * NO_; i += 256) WcB[i] = Wc[i];
  const float* W1g = W1 + (size_t)s * NO_ * H_;
  for (int i = tid; i < NO_ * HP_; i += 256) {
    int no = i / HP_, h = i % HP_;
    float v = (h < H_) ? W1g[no * H_ + h] : 0.0f;
    _Float16 hv = (_Float16)v;
    W1hB[no * HP_ + h] = hv;
    W1tB[h * NO_ + no] = hv;
  }
  for (int i = tid; i < HP_; i += 256) {
    b1B[i] = (i < H_) ? b1[s * H_ + i] : 0.0f;
    W2B[i] = (i < H_) ? W2[s * H_ + i] : 0.0f;
  }
  for (int i = tid; i < MT_ * (HP_ - HT_); i += 256) {
    int a = i / (HP_ - HT_), c = HT_ + i % (HP_ - HT_);
    gzB[a * HP_ + c] = (_Float16)0.0f;
  }
  __syncthreads();

  // ---- P3: feat = sin(pos@Wd + cell@Wc), keep cos(u) for backward ----
  for (int i = tid; i < MT_ * NO_; i += 256) {
    int a = i >> 7, j = i & 127;
    float u = 0.0f;
#pragma unroll
    for (int q = 0; q < 9; ++q) u += cellB[a * 9 + q] * WcB[q * NO_ + j];
#pragma unroll
    for (int k = 0; k < 3; ++k) u += posB[a * 3 + k] * WdB[k * NO_ + j];
    float sn, cs;
    __sincosf(u, &sn, &cs);
    featB[a * NO_ + j] = (_Float16)sn;
    cosB [a * NO_ + j] = (_Float16)cs;
  }
  __syncthreads();

  // ---- P4: per-wave WMMA pipeline (16 atoms per wave) ----
  const int wv    = tid >> 5;
  const int lane  = tid & 31;
  const int laneN = lane & 15;      // fragment column
  const int halfw = lane >> 4;      // lane-half per ISA 16-bit A/B layouts
  const int lo8   = halfw * 8;
  const int lo16  = halfw * 16;
  const int m0    = wv * 16;
  const int arow  = m0 + laneN;     // A-matrix row held by this lane

  // A fragments for forward GEMM (K = 128, 4 ktiles of 32), loaded once.
  v16h aF[4];
#pragma unroll
  for (int kt = 0; kt < 4; ++kt) {
    const _Float16* ap = &featB[arow * NO_ + kt * 32];
#pragma unroll
    for (int i = 0; i < 16; ++i) aF[kt][i] = ap[lo16 + i];
  }

  float esAcc[8];
#pragma unroll
  for (int r = 0; r < 8; ++r) esAcc[r] = 0.0f;

  // Forward: z[16 x 176] = feat @ W1[s]; fused silu / W2-dot / gz epilogue.
  for (int nt = 0; nt < HT_ / 16; ++nt) {
    // batch all B-fragment loads first so ds_load_b128s overlap the WMMA chain
    v16h bF[4];
#pragma unroll
    for (int kt = 0; kt < 4; ++kt) {
      const _Float16* bp = &W1tB[(nt * 16 + laneN) * NO_ + kt * 32 + lo16];
#pragma unroll
      for (int i = 0; i < 16; ++i) bF[kt][i] = bp[i];
    }
    v8f c = {};
#pragma unroll
    for (int kt = 0; kt < 4; ++kt)
      c = __builtin_amdgcn_wmma_f32_16x16x32_f16(false, aF[kt], false, bF[kt],
                                                 (short)0, c, false, false);

    const int col = nt * 16 + laneN;
    const float b1v = b1B[col];
    const float w2v = W2B[col];
#pragma unroll
    for (int r = 0; r < 8; ++r) {               // C layout: VGPR r -> row r (+8 hi half)
      float z   = c[r] + b1v;
      float sig = fast_sigmoid(z);
      esAcc[r] += (z * sig) * w2v;              // silu(z) * W2
      float ds  = sig * (1.0f + z * (1.0f - sig));   // silu'(z)
      int row   = m0 + r + halfw * 8;
      gzB[row * HP_ + col] = (_Float16)(w2v * ds);
    }
  }

  // Backward: gf[16 x 128] = gz @ W1[s]^T ; A-fragments are ntile-invariant ->
  // preload all 6 once (own-wave gz rows; ds waits ordered by compiler).
  v16h aB[HP_ / 32];
#pragma unroll
  for (int kt2 = 0; kt2 < HP_ / 32; ++kt2) {
    const _Float16* ap = &gzB[arow * HP_ + kt2 * 32];
#pragma unroll
    for (int i = 0; i < 16; ++i) aB[kt2][i] = ap[lo16 + i];
  }

  float fAcc[8][3];
#pragma unroll
  for (int r = 0; r < 8; ++r) { fAcc[r][0] = fAcc[r][1] = fAcc[r][2] = 0.0f; }

  for (int nt2 = 0; nt2 < NO_ / 16; ++nt2) {
    v16h bB[HP_ / 32];
#pragma unroll
    for (int kt2 = 0; kt2 < HP_ / 32; ++kt2) {
      const _Float16* bp = &W1hB[(nt2 * 16 + laneN) * HP_ + kt2 * 32 + lo16];
#pragma unroll
      for (int i = 0; i < 16; ++i) bB[kt2][i] = bp[i];
    }
    v8f c2 = {};
#pragma unroll
    for (int kt2 = 0; kt2 < HP_ / 32; ++kt2)
      c2 = __builtin_amdgcn_wmma_f32_16x16x32_f16(false, aB[kt2], false, bB[kt2],
                                                  (short)0, c2, false, false);

    const int j = nt2 * 16 + laneN;
    const float wd0 = WdB[j], wd1 = WdB[NO_ + j], wd2 = WdB[2 * NO_ + j];
#pragma unroll
    for (int r = 0; r < 8; ++r) {
      int row = m0 + r + halfw * 8;
      float g = c2[r] * (float)cosB[row * NO_ + j];
      fAcc[r][0] += wd0 * g; fAcc[r][1] += wd1 * g; fAcc[r][2] += wd2 * g;
    }
  }

  // ---- reductions across the 16-lane halves + writes ----
  const float b2v = b2[s];
#pragma unroll
  for (int r = 0; r < 8; ++r) {
    float e  = esAcc[r];
    float f0 = fAcc[r][0], f1 = fAcc[r][1], f2 = fAcc[r][2];
#pragma unroll
    for (int m = 1; m < 16; m <<= 1) {
      e  += __shfl_xor(e,  m, 32);
      f0 += __shfl_xor(f0, m, 32);
      f1 += __shfl_xor(f1, m, 32);
      f2 += __shfl_xor(f2, m, 32);
    }
    if (laneN == 0) {                           // lane 0 -> row r, lane 16 -> row r+8
      int row = m0 + r + halfw * 8;
      int ai  = aidxB[row];
      if (ai >= 0) {
        float es = e + b2v;
        __builtin_nontemporal_store(es, &energies[ai]);
        __builtin_nontemporal_store(f0, &forces[ai * 3 + 0]);
        __builtin_nontemporal_store(f1, &forces[ai * 3 + 1]);
        __builtin_nontemporal_store(f2, &forces[ai * 3 + 2]);
        long long q = __double2ll_rn((double)es * 4294967296.0);
        atomicAdd(&esum[cidxB[row]], (unsigned long long)q);
      }
    }
  }
}

// ---------------- host-side launch ----------------
extern "C" void kernel_launch(void* const* d_in, const int* in_sizes, int n_in,
                              void* d_out, int out_size, void* d_ws, size_t ws_size,
                              hipStream_t stream) {
  const int*   symbols    = (const int*)d_in[0];
  const float* positions  = (const float*)d_in[1];
  const float* cells      = (const float*)d_in[2];
  const int*   crystalidx = (const int*)d_in[3];
  /* d_in[4] = pbcs (unused by the stand-in descriptor) */
  const float* Wd = (const float*)d_in[5];
  const float* Wc = (const float*)d_in[6];
  const float* W1 = (const float*)d_in[7];
  const float* b1 = (const float*)d_in[8];
  const float* W2 = (const float*)d_in[9];
  const float* b2 = (const float*)d_in[10];

  float* energies = (float*)d_out;          // [NTA]
  float* energy   = energies + NTA_;        // [NC]
  float* forces   = energy + NC_;           // [NTA,3]

  // workspace layout (256 + NC*8 + NTA*4 ~= 2.03 MB)
  char* ws = (char*)d_ws;
  int* counts    = (int*)ws;                // [4]
  int* offsets   = counts + 8;              // [5]
  int* cursors   = counts + 16;             // [4]
  int* tileStart = counts + 24;             // [5]
  unsigned long long* esum = (unsigned long long*)(ws + 256);      // [NC]
  int* sortedIdx = (int*)(ws + 256 + NC_ * 8);                     // [NTA]

  const int nb = (NTA_ + 255) / 256;
  const int maxTiles = (NTA_ + MT_ - 1) / MT_ + S_;

  k_init   <<<(NC_ + 255) / 256, 256, 0, stream>>>(counts, esum);
  k_count  <<<nb, 256, 0, stream>>>(symbols, counts);
  k_scan   <<<1, 32, 0, stream>>>(counts, offsets, cursors, tileStart);
  k_scatter<<<nb, 256, 0, stream>>>(symbols, cursors, sortedIdx);

  (void)hipFuncSetAttribute((const void*)k_main,
                            hipFuncAttributeMaxDynamicSharedMemorySize, SMEM_BYTES);
  k_main<<<maxTiles, 256, SMEM_BYTES, stream>>>(
      positions, cells, crystalidx, Wd, Wc, W1, b1, W2, b2,
      offsets, tileStart, sortedIdx, energies, esum, forces);

  k_final<<<(NC_ + 255) / 256, 256, 0, stream>>>(esum, energy);

  (void)in_sizes; (void)n_in; (void)out_size; (void)ws_size;
}
